// SwinTransformerBlock_13116830122338
// MI455X (gfx1250) — matrix operand order repaired and checked by reading, not compiled, not run on hardware
//
#include <hip/hip_runtime.h>

// ---------------------------------------------------------------------------
// Swin Transformer block for MI455X (gfx1250), wave32 + WMMA bf16.
// B=8, H=W=128, C=192, HEADS=6, WS=8, SS=4, MLP_H=384, N=64, NW=256
// ---------------------------------------------------------------------------

typedef __bf16 bf16;
typedef __attribute__((ext_vector_type(16))) __bf16 v16bf;
typedef __attribute__((ext_vector_type(8)))  float   v8f;
typedef __attribute__((ext_vector_type(4)))  int     v4i;

#define CDIM   192
#define HEADS  6
#define HD     32
#define WS     8
#define SS     4
#define NWIN   2048          // B * NW
#define NWIMG  256           // windows per image
#define MTOT   131072        // B*H*W tokens
#define MLPH   384

#if __has_builtin(__builtin_amdgcn_global_load_async_to_lds_b128) && \
    __has_builtin(__builtin_amdgcn_s_wait_asynccnt)
#define USE_ASYNC_LDS 1
#else
#define USE_ASYNC_LDS 0
#endif

__device__ __forceinline__ bf16 f2bf(float f) { return (bf16)f; }

__device__ __forceinline__ v8f wmma_bf16(v16bf a, v16bf b, v8f c) {
  return __builtin_amdgcn_wmma_f32_16x16x32_bf16(
      /*neg_a=*/false, a, /*neg_b=*/false, b,
      /*c_mod=*/(short)0, c, /*reuse_a=*/false, /*reuse_b=*/false);
}

// A-matrix fragment, 16x32 bf16 (MxK), row-major source with leading dim lda.
// lane<16 -> M=lane, K in {0..7,16..23}; lane>=16 -> M=lane-16, K {8..15,24..31}
__device__ __forceinline__ v16bf load_a_frag(const bf16* __restrict__ A, int lda) {
  int lane = threadIdx.x & 31;
  int row  = lane & 15;
  int kb   = (lane >> 4) * 8;
  const bf16* p = A + (size_t)row * lda + kb;
  v16bf f;
#pragma unroll
  for (int e = 0; e < 8; ++e) f[e] = p[e];
#pragma unroll
  for (int e = 0; e < 8; ++e) f[8 + e] = p[16 + e];
  return f;
}

// B-matrix fragment, 32x16 bf16 (KxN) from Wt = B^T stored row-major [N, K].
// lanes 0..15: column N=lane, K {0..7,16..23}; lanes 16..31: column N=lane-16, K {8..15,24..31}
__device__ __forceinline__ v16bf load_b_frag(const bf16* __restrict__ Wt, int ldw) {
  int lane = threadIdx.x & 31;
  int col  = lane & 15;
  int kb   = (lane >> 4) * 8;
  const bf16* p = Wt + (size_t)col * ldw + kb;
  v16bf f;
#pragma unroll
  for (int e = 0; e < 8; ++e) { f[e] = p[e]; f[8 + e] = p[16 + e]; }
  return f;
}

// ---------------------------------------------------------------------------
// Elementwise f32 -> bf16 (weights)
// ---------------------------------------------------------------------------
__global__ void f32_to_bf16_kernel(const float* __restrict__ in,
                                   bf16* __restrict__ out, int n) {
  int i = blockIdx.x * 256 + threadIdx.x;
  if (i < n) out[i] = f2bf(in[i]);
}

// ---------------------------------------------------------------------------
// LN1 + cyclic shift (-SS,-SS) + window partition, fp32 -> bf16
// ---------------------------------------------------------------------------
__global__ __launch_bounds__(256)
void ln_window_kernel(const float* __restrict__ x, const float* __restrict__ g,
                      const float* __restrict__ bb, bf16* __restrict__ out) {
  int T    = blockIdx.x * 8 + (threadIdx.x >> 5);
  int lane = threadIdx.x & 31;
  int wi = T >> 6, n = T & 63;
  int b  = wi >> 8, rem = wi & 255;
  int wh = rem >> 4, ww = rem & 15;
  int i  = n >> 3,  j  = n & 7;
  int ph = (wh * WS + i + SS) & 127;   // roll(-SS) source pixel
  int pw = (ww * WS + j + SS) & 127;
  const float* src = x + ((size_t)b * 16384 + (size_t)ph * 128 + pw) * CDIM;

  float v[6];
  float s = 0.f;
#pragma unroll
  for (int k = 0; k < 6; ++k) { v[k] = src[lane + 32 * k]; s += v[k]; }
#pragma unroll
  for (int off = 16; off; off >>= 1) s += __shfl_xor(s, off, 32);
  float mean = s * (1.0f / CDIM);
  float vs = 0.f;
#pragma unroll
  for (int k = 0; k < 6; ++k) { float d = v[k] - mean; vs += d * d; }
#pragma unroll
  for (int off = 16; off; off >>= 1) vs += __shfl_xor(vs, off, 32);
  float rstd = rsqrtf(vs * (1.0f / CDIM) + 1e-5f);

  bf16* dst = out + (size_t)T * CDIM;
#pragma unroll
  for (int k = 0; k < 6; ++k) {
    int c = lane + 32 * k;
    dst[c] = f2bf((v[k] - mean) * rstd * g[c] + bb[c]);
  }
}

// ---------------------------------------------------------------------------
// LN2 (plain): x1 f32 -> bf16
// ---------------------------------------------------------------------------
__global__ __launch_bounds__(256)
void ln_plain_kernel(const float* __restrict__ x, const float* __restrict__ g,
                     const float* __restrict__ bb, bf16* __restrict__ out) {
  int T    = blockIdx.x * 8 + (threadIdx.x >> 5);
  int lane = threadIdx.x & 31;
  const float* src = x + (size_t)T * CDIM;
  float v[6];
  float s = 0.f;
#pragma unroll
  for (int k = 0; k < 6; ++k) { v[k] = src[lane + 32 * k]; s += v[k]; }
#pragma unroll
  for (int off = 16; off; off >>= 1) s += __shfl_xor(s, off, 32);
  float mean = s * (1.0f / CDIM);
  float vs = 0.f;
#pragma unroll
  for (int k = 0; k < 6; ++k) { float d = v[k] - mean; vs += d * d; }
#pragma unroll
  for (int off = 16; off; off >>= 1) vs += __shfl_xor(vs, off, 32);
  float rstd = rsqrtf(vs * (1.0f / CDIM) + 1e-5f);
  bf16* dst = out + (size_t)T * CDIM;
#pragma unroll
  for (int k = 0; k < 6; ++k) {
    int c = lane + 32 * k;
    dst[c] = f2bf((v[k] - mean) * rstd * g[c] + bb[c]);
  }
}

// ---------------------------------------------------------------------------
// Stage 64 contiguous weight rows (64*K bf16) into LDS.
// Async global->LDS (ASYNCcnt) path on gfx1250, else vectorized copy.
// ---------------------------------------------------------------------------
template <int KMAX>
__device__ __forceinline__ void stage_weights(const bf16* __restrict__ wsrc,
                                              bf16* __restrict__ lds, int K,
                                              int nthreads) {
  const int nbytes = 64 * K * (int)sizeof(bf16);
#if USE_ASYNC_LDS
  for (int off = threadIdx.x * 16; off < nbytes; off += nthreads * 16) {
    __builtin_amdgcn_global_load_async_to_lds_b128(
        (v4i*)((const char*)wsrc + off), (v4i*)((char*)lds + off), 0, 0);
  }
  __builtin_amdgcn_s_wait_asynccnt(0);
  __syncthreads();
#else
  const uint4* s = (const uint4*)wsrc;
  uint4* d = (uint4*)lds;
  const int nchunks = nbytes / 16;
  for (int i = threadIdx.x; i < nchunks; i += nthreads) d[i] = s[i];
  __syncthreads();
#endif
}

// ---------------------------------------------------------------------------
// Generic GEMM: out[M,N] = act(A[M,K] @ W[N,K]^T + bias)
// MODE 0: bf16 store; MODE 1: exact GELU, bf16 store;
// MODE 2: f32 store with residual: out_f = res + acc + bias
// Block: 128 threads = 4 waves; block tile 128x64, wave tile 32x64.
// Weight tile (64 rows x K) staged in LDS once per block.
// Grid: (M/128, N/64)
// ---------------------------------------------------------------------------
template <int MODE, int KMAX>
__global__ __launch_bounds__(128)
void gemm_bf16_wmma(const bf16* __restrict__ A, const bf16* __restrict__ W,
                    const float* __restrict__ bias, const float* __restrict__ res,
                    bf16* __restrict__ out_bf, float* __restrict__ out_f,
                    int M, int K, int N) {
  __shared__ __align__(16) bf16 Bl[64 * KMAX];
  const int lane = threadIdx.x & 31;
  const int wave = threadIdx.x >> 5;
  const int m0 = blockIdx.x * 128 + wave * 32;
  const int n0 = blockIdx.y * 64;

  stage_weights<KMAX>(W + (size_t)n0 * K, Bl, K, 128);

  v8f z = {};
  v8f acc0[4], acc1[4];
#pragma unroll
  for (int t = 0; t < 4; ++t) { acc0[t] = z; acc1[t] = z; }

  for (int kk = 0; kk < K; kk += 32) {
    v16bf a0 = load_a_frag(A + (size_t)m0 * K + kk, K);
    v16bf a1 = load_a_frag(A + (size_t)(m0 + 16) * K + kk, K);
#pragma unroll
    for (int tn = 0; tn < 4; ++tn) {
      v16bf b = load_b_frag(Bl + (size_t)(tn * 16) * K + kk, K);
      acc0[tn] = wmma_bf16(a0, b, acc0[tn]);
      acc1[tn] = wmma_bf16(a1, b, acc1[tn]);
    }
  }

  const int rowadd = (lane >> 4) * 8;
  const int coll   = lane & 15;
#pragma unroll
  for (int half = 0; half < 2; ++half) {
    int mh = m0 + half * 16;
#pragma unroll
    for (int tn = 0; tn < 4; ++tn) {
      int col = n0 + tn * 16 + coll;
      float bv = bias[col];
      v8f acc = half ? acc1[tn] : acc0[tn];
#pragma unroll
      for (int r = 0; r < 8; ++r) {
        int row = mh + r + rowadd;
        size_t o = (size_t)row * N + col;
        float vacc = acc[r] + bv;
        if (MODE == 0) {
          out_bf[o] = f2bf(vacc);
        } else if (MODE == 1) {
          float gl = 0.5f * vacc * (1.0f + erff(vacc * 0.70710678118654752f));
          out_bf[o] = f2bf(gl);
        } else {
          out_f[o] = res[o] + vacc;
        }
      }
    }
  }
}

// window-token index -> pixel row index after window-reverse + roll(+SS)
__device__ __forceinline__ int win_to_pix(int t) {
  int wi = t >> 6, n = t & 63;
  int b  = wi >> 8, rem = wi & 255;
  int wh = rem >> 4, ww = rem & 15;
  int i  = n >> 3,  j  = n & 7;
  int p = (wh * WS + i + SS) & 127;
  int q = (ww * WS + j + SS) & 127;
  return (b << 14) + (p << 7) + q;
}

// ---------------------------------------------------------------------------
// Proj GEMM + window reverse + roll + residual add:
// x1[pix,:] = x[pix,:] + attn_out[t,:] @ proj_w^T + proj_b
// ---------------------------------------------------------------------------
__global__ __launch_bounds__(128)
void proj_gemm_kernel(const bf16* __restrict__ A, const bf16* __restrict__ W,
                      const float* __restrict__ bias, const float* __restrict__ xres,
                      float* __restrict__ out) {
  const int K = CDIM, N = CDIM;
  __shared__ __align__(16) bf16 Bl[64 * CDIM];
  const int lane = threadIdx.x & 31;
  const int wave = threadIdx.x >> 5;
  const int m0 = blockIdx.x * 128 + wave * 32;
  const int n0 = blockIdx.y * 64;

  stage_weights<CDIM>(W + (size_t)n0 * K, Bl, K, 128);

  v8f z = {};
  v8f acc0[4], acc1[4];
#pragma unroll
  for (int t = 0; t < 4; ++t) { acc0[t] = z; acc1[t] = z; }

  for (int kk = 0; kk < K; kk += 32) {
    v16bf a0 = load_a_frag(A + (size_t)m0 * K + kk, K);
    v16bf a1 = load_a_frag(A + (size_t)(m0 + 16) * K + kk, K);
#pragma unroll
    for (int tn = 0; tn < 4; ++tn) {
      v16bf b = load_b_frag(Bl + (size_t)(tn * 16) * K + kk, K);
      acc0[tn] = wmma_bf16(a0, b, acc0[tn]);
      acc1[tn] = wmma_bf16(a1, b, acc1[tn]);
    }
  }

  const int rowadd = (lane >> 4) * 8;
  const int coll   = lane & 15;
#pragma unroll
  for (int half = 0; half < 2; ++half) {
    int mh = m0 + half * 16;
#pragma unroll
    for (int tn = 0; tn < 4; ++tn) {
      int col = n0 + tn * 16 + coll;
      float bv = bias[col];
      v8f acc = half ? acc1[tn] : acc0[tn];
#pragma unroll
      for (int r = 0; r < 8; ++r) {
        int t = mh + r + rowadd;
        int pix = win_to_pix(t);
        size_t o = (size_t)pix * N + col;
        out[o] = xres[o] + acc[r] + bv;
      }
    }
  }
}

// ---------------------------------------------------------------------------
// Windowed attention: one wave per (window, head).
// qkv layout: [NWIN*64, 576]; q at col h*32, k at 192+h*32, v at 384+h*32.
// ---------------------------------------------------------------------------
__global__ __launch_bounds__(32)
void attn_kernel(const bf16* __restrict__ qkv, const float* __restrict__ rpb,
                 const int* __restrict__ rel_idx, const float* __restrict__ mask,
                 bf16* __restrict__ out) {
  __shared__ float Sbuf[64 * 64];
  __shared__ bf16  Pbuf[64 * 64];
  __shared__ bf16  vT[32 * 64];

  const int wh = blockIdx.x;
  const int w  = wh / HEADS;
  const int h  = wh - w * HEADS;
  const int lane = threadIdx.x;

  const bf16* base = qkv + (size_t)w * 64 * 576;
  const bf16* qp = base + h * HD;
  const bf16* kp = base + CDIM + h * HD;
  const bf16* vp = base + 2 * CDIM + h * HD;

  // stage V transposed: vT[d][m] = v[m][d]
  for (int idx = lane; idx < 2048; idx += 32) {
    int m = idx >> 5, d = idx & 31;
    vT[d * 64 + m] = vp[(size_t)m * 576 + d];
  }

  // ---- S = q @ k^T (K=32, one WMMA k-step) ----
  v16bf qa[4];
#pragma unroll
  for (int tm = 0; tm < 4; ++tm) qa[tm] = load_a_frag(qp + (size_t)tm * 16 * 576, 576);

  const int rowadd = (lane >> 4) * 8;
  const int coll   = lane & 15;
#pragma unroll
  for (int tn = 0; tn < 4; ++tn) {
    v16bf kb = load_b_frag(kp + (size_t)tn * 16 * 576, 576);
#pragma unroll
    for (int tm = 0; tm < 4; ++tm) {
      v8f c = {};
      c = wmma_bf16(qa[tm], kb, c);
#pragma unroll
      for (int r = 0; r < 8; ++r) {
        int row = tm * 16 + r + rowadd;
        int col = tn * 16 + coll;
        Sbuf[row * 64 + col] = c[r];
      }
    }
  }
  __syncthreads();

  // ---- softmax with scale + relative position bias + shift mask ----
  const float scale = 0.1767766952966369f;  // 32^-0.5
  const int nw = w & (NWIMG - 1);
#pragma unroll
  for (int rr = 0; rr < 2; ++rr) {
    int r = lane + rr * 32;
    float logits[64];
    float mx = -1e30f;
    for (int m = 0; m < 64; ++m) {
      float bias = rpb[rel_idx[r * 64 + m] * HEADS + h];
      float mk   = mask[(size_t)nw * 4096 + r * 64 + m];
      float vv   = Sbuf[r * 64 + m] * scale + bias + mk;
      logits[m] = vv;
      mx = fmaxf(mx, vv);
    }
    float ssum = 0.f;
    for (int m = 0; m < 64; ++m) {
      float e = __expf(logits[m] - mx);
      logits[m] = e;
      ssum += e;
    }
    float inv = 1.0f / ssum;
    for (int m = 0; m < 64; ++m) Pbuf[r * 64 + m] = f2bf(logits[m] * inv);
  }
  __syncthreads();

  // ---- O = P @ V : M=64, N=32, K=64 (2 k-steps) ----
  v8f o[4][2];
  v8f z = {};
#pragma unroll
  for (int tm = 0; tm < 4; ++tm)
#pragma unroll
    for (int tn = 0; tn < 2; ++tn) o[tm][tn] = z;

#pragma unroll
  for (int kk = 0; kk < 2; ++kk) {
    v16bf pa[4];
#pragma unroll
    for (int tm = 0; tm < 4; ++tm)
      pa[tm] = load_a_frag(Pbuf + tm * 16 * 64 + kk * 32, 64);
#pragma unroll
    for (int tn = 0; tn < 2; ++tn) {
      v16bf vb = load_b_frag(vT + tn * 16 * 64 + kk * 32, 64);
#pragma unroll
      for (int tm = 0; tm < 4; ++tm) o[tm][tn] = wmma_bf16(pa[tm], vb, o[tm][tn]);
    }
  }

  // store O -> attn_out[w*64+row, h*32 + col], bf16
#pragma unroll
  for (int tm = 0; tm < 4; ++tm) {
#pragma unroll
    for (int tn = 0; tn < 2; ++tn) {
#pragma unroll
      for (int r = 0; r < 8; ++r) {
        int row = tm * 16 + r + rowadd;
        int col = tn * 16 + coll;
        out[(size_t)(w * 64 + row) * CDIM + h * HD + col] = f2bf(o[tm][tn][r]);
      }
    }
  }
}

// ---------------------------------------------------------------------------
// Host launch
// ---------------------------------------------------------------------------
extern "C" void kernel_launch(void* const* d_in, const int* in_sizes, int n_in,
                              void* d_out, int out_size, void* d_ws, size_t ws_size,
                              hipStream_t stream) {
  (void)in_sizes; (void)n_in; (void)out_size; (void)ws_size;

  const float* x        = (const float*)d_in[0];
  const float* n1g      = (const float*)d_in[1];
  const float* n1b      = (const float*)d_in[2];
  const float* qkv_w    = (const float*)d_in[3];
  const float* qkv_b    = (const float*)d_in[4];
  const float* rpb      = (const float*)d_in[5];
  const float* proj_w   = (const float*)d_in[6];
  const float* proj_b   = (const float*)d_in[7];
  const float* n2g      = (const float*)d_in[8];
  const float* n2b      = (const float*)d_in[9];
  const float* fc1_w    = (const float*)d_in[10];
  const float* fc1_b    = (const float*)d_in[11];
  const float* fc2_w    = (const float*)d_in[12];
  const float* fc2_b    = (const float*)d_in[13];
  const int*   rel_idx  = (const int*)d_in[14];
  const float* attn_mask= (const float*)d_in[15];
  float* out = (float*)d_out;

  // --- workspace layout (region reuse) ---
  char* ws = (char*)d_ws;
  bf16* qkvw_bf = (bf16*)ws;                       // 576*192
  bf16* projw_bf = qkvw_bf + 576 * 192;            // 192*192
  bf16* fc1w_bf  = projw_bf + 192 * 192;           // 384*192
  bf16* fc2w_bf  = fc1w_bf + 384 * 192;            // 192*384
  size_t offB = (((size_t)(fc2w_bf + 192 * 384)) + 255) & ~(size_t)255;
  // Region B (50.3 MB bf16): xw -> attn_out -> h2 (sequential lifetimes)
  bf16* regionB = (bf16*)offB;
  size_t offA = (offB + (size_t)MTOT * CDIM * sizeof(bf16) + 255) & ~(size_t)255;
  // Region A (151 MB): qkv activations (bf16), later x1 (f32, 100 MB)
  bf16*  qkv_act = (bf16*)offA;
  float* x1      = (float*)offA;
  size_t offC = (offA + (size_t)MTOT * 576 * sizeof(bf16) + 255) & ~(size_t)255;
  // Region C (100.7 MB): mlp hidden bf16
  bf16* mlp1 = (bf16*)offC;

  // --- weight conversions ---
  f32_to_bf16_kernel<<<(576 * 192 + 255) / 256, 256, 0, stream>>>(qkv_w, qkvw_bf, 576 * 192);
  f32_to_bf16_kernel<<<(192 * 192 + 255) / 256, 256, 0, stream>>>(proj_w, projw_bf, 192 * 192);
  f32_to_bf16_kernel<<<(384 * 192 + 255) / 256, 256, 0, stream>>>(fc1_w, fc1w_bf, 384 * 192);
  f32_to_bf16_kernel<<<(192 * 384 + 255) / 256, 256, 0, stream>>>(fc2_w, fc2w_bf, 192 * 384);

  // --- LN1 + shift + window partition ---
  ln_window_kernel<<<MTOT / 8, 256, 0, stream>>>(x, n1g, n1b, regionB);

  // --- QKV GEMM: [M,192] x [576,192]^T -> bf16 [M,576] ---
  gemm_bf16_wmma<0, CDIM><<<dim3(MTOT / 128, 576 / 64), 128, 0, stream>>>(
      regionB, qkvw_bf, qkv_b, nullptr, qkv_act, nullptr, MTOT, CDIM, 576);

  // --- attention: one wave per (window, head) ---
  attn_kernel<<<NWIN * HEADS, 32, 0, stream>>>(qkv_act, rpb, rel_idx, attn_mask, regionB);

  // --- proj GEMM + window reverse + roll + residual -> x1 (f32) ---
  proj_gemm_kernel<<<dim3(MTOT / 128, CDIM / 64), 128, 0, stream>>>(
      regionB, projw_bf, proj_b, x, x1);

  // --- LN2 -> bf16 ---
  ln_plain_kernel<<<MTOT / 8, 256, 0, stream>>>(x1, n2g, n2b, regionB);

  // --- fc1 GEMM + GELU -> bf16 [M,384] ---
  gemm_bf16_wmma<1, CDIM><<<dim3(MTOT / 128, MLPH / 64), 128, 0, stream>>>(
      regionB, fc1w_bf, fc1_b, nullptr, mlp1, nullptr, MTOT, CDIM, MLPH);

  // --- fc2 GEMM + residual -> d_out (f32) ---
  gemm_bf16_wmma<2, MLPH><<<dim3(MTOT / 128, CDIM / 64), 128, 0, stream>>>(
      mlp1, fc2w_bf, fc2_b, x1, nullptr, out, MTOT, MLPH, CDIM);
}
